// NonLinearAttention_6339371729512
// MI455X (gfx1250) — matrix operand order repaired and checked by reading, not compiled, not run on hardware
//
#include <hip/hip_runtime.h>
#include <hip/hip_bf16.h>
#include <cstdint>

// ---------------------------------------------------------------------------
// Types for CDNA5 WMMA bf16: D(f32 16x16) = A(bf16 16x32) x B(bf16 32x16) + C
// ---------------------------------------------------------------------------
typedef __attribute__((ext_vector_type(16))) __bf16 v16bf;
typedef __attribute__((ext_vector_type(8)))  float  v8f;

union BF16Frag { v16bf v; uint4 q[2]; };

__device__ __forceinline__ unsigned short f2bf(float f) {
  unsigned int u = __float_as_uint(f);
  u += 0x7FFFu + ((u >> 16) & 1u);          // round-to-nearest-even
  return (unsigned short)(u >> 16);
}
__device__ __forceinline__ float bf2f(unsigned short h) {
  return __uint_as_float(((unsigned int)h) << 16);
}

// ---------------------------------------------------------------------------
// Weight convert + transpose: W[K,N] f32  ->  Wt[N,K] bf16
// (makes every WMMA B-fragment two contiguous 16-byte runs per lane)
// ---------------------------------------------------------------------------
__global__ __launch_bounds__(256) void wconv_kernel(
    const float* __restrict__ W, unsigned short* __restrict__ Wt, int K, int N) {
  int idx = blockIdx.x * 256 + threadIdx.x;
  if (idx < K * N) {
    int k = idx / N, n = idx % N;
    Wt[(size_t)n * K + k] = f2bf(W[idx]);
  }
}

// ---------------------------------------------------------------------------
// Fused 2-layer MLP: out = (relu(X @ W1 + b1)) @ W2 + b2
//   X rows staged in LDS as bf16; hidden kept in LDS (never touches HBM).
//   Weights are pre-transposed bf16 [Ncols, K].
//   X may be a concat of two inputs (X0: fin0 cols, X1: FIN-fin0 cols).
//   64 rows / block, 8 waves; stage1: wave owns 4 coltiles x 4 rowtiles,
//   stage2: 2 coltiles x 4 rowtiles.
// ---------------------------------------------------------------------------
template <int FIN, int HID, int FOUT, bool OUT_BF16>
__global__ __launch_bounds__(256) void mlp_kernel(
    const float* __restrict__ X0, const float* __restrict__ X1, int fin0,
    const unsigned short* __restrict__ W1t, const float* __restrict__ B1,
    const unsigned short* __restrict__ W2t, const float* __restrict__ B2,
    void* __restrict__ outp) {
  constexpr int ROWS = 64;
  constexpr int RT   = ROWS / 16;        // 4 row tiles
  constexpr int CPW1 = (HID / 16) / 8;   // col tiles per wave, stage 1 (=4)
  constexpr int CPW2 = (FOUT / 16) / 8;  // col tiles per wave, stage 2 (=2)

  extern __shared__ __align__(16) unsigned short lds[];
  unsigned short* Xs = lds;                  // [ROWS][FIN]  bf16
  unsigned short* Hs = lds + ROWS * FIN;     // [ROWS][HID]  bf16

  const int tid    = threadIdx.x;
  const int lane   = tid & 31;
  const int wave   = tid >> 5;
  const int laneLo = lane & 15;
  const int laneHi = lane >> 4;
  const long long rowbase = (long long)blockIdx.x * ROWS;

  // ---- stage 0: load f32 inputs, convert to bf16 in LDS ----
  const int stride1 = FIN - fin0;
  for (int i = tid * 4; i < ROWS * FIN; i += 256 * 4) {
    int r = i / FIN, c = i % FIN;
    float4 f;
    if (c < fin0)
      f = *(const float4*)(X0 + (rowbase + r) * (long long)fin0 + c);
    else
      f = *(const float4*)(X1 + (rowbase + r) * (long long)stride1 + (c - fin0));
    Xs[i + 0] = f2bf(f.x); Xs[i + 1] = f2bf(f.y);
    Xs[i + 2] = f2bf(f.z); Xs[i + 3] = f2bf(f.w);
  }
  __syncthreads();

  const v8f vzero = {0.f, 0.f, 0.f, 0.f, 0.f, 0.f, 0.f, 0.f};

  // ---- stage 1: H = relu(X @ W1 + b1), kept in LDS as bf16 ----
  {
    v8f acc[RT][CPW1];
    for (int i = 0; i < RT; ++i)
      for (int j = 0; j < CPW1; ++j) acc[i][j] = vzero;

    for (int k0 = 0; k0 < FIN; k0 += 32) {
      BF16Frag afr[RT];
#pragma unroll
      for (int rt = 0; rt < RT; ++rt) {
        // A 16-bit 16x32 layout: lanes 0-15 K=0..7 & 16..23, lanes 16-31 K=8..15 & 24..31
        const unsigned short* p = Xs + (rt * 16 + laneLo) * FIN + k0 + laneHi * 8;
        afr[rt].q[0] = *(const uint4*)(p);
        afr[rt].q[1] = *(const uint4*)(p + 16);
      }
#pragma unroll
      for (int ct = 0; ct < CPW1; ++ct) {
        const int col = (wave * CPW1 + ct) * 16 + laneLo;
        // B 16-bit 32x16 layout: lanes 0-15 K=0..7 & 16..23, lanes 16-31 K=8..15 & 24..31
        const unsigned short* p = W1t + (size_t)col * FIN + k0 + laneHi * 8;
        BF16Frag bfr;
        bfr.q[0] = *(const uint4*)(p);
        bfr.q[1] = *(const uint4*)(p + 16);
        __builtin_prefetch(p + 32, 0, 1);
#pragma unroll
        for (int rt = 0; rt < RT; ++rt)
          acc[rt][ct] = __builtin_amdgcn_wmma_f32_16x16x32_bf16(
              false, afr[rt].v, false, bfr.v, (short)0, acc[rt][ct], false, false);
      }
    }
#pragma unroll
    for (int ct = 0; ct < CPW1; ++ct) {
      const int col  = (wave * CPW1 + ct) * 16 + laneLo;
      const float bs = B1[col];
#pragma unroll
      for (int rt = 0; rt < RT; ++rt)
#pragma unroll
        for (int r = 0; r < 8; ++r) {
          const int row = rt * 16 + r + laneHi * 8;  // C/D layout: M = r + 8*(lane>=16)
          float h = acc[rt][ct][r] + bs;
          Hs[row * HID + col] = f2bf(fmaxf(h, 0.f));
        }
    }
  }
  __syncthreads();

  // ---- stage 2: out = H @ W2 + b2 ----
  {
    v8f acc[RT][CPW2];
    for (int i = 0; i < RT; ++i)
      for (int j = 0; j < CPW2; ++j) acc[i][j] = vzero;

    for (int k0 = 0; k0 < HID; k0 += 32) {
      BF16Frag afr[RT];
#pragma unroll
      for (int rt = 0; rt < RT; ++rt) {
        const unsigned short* p = Hs + (rt * 16 + laneLo) * HID + k0 + laneHi * 8;
        afr[rt].q[0] = *(const uint4*)(p);
        afr[rt].q[1] = *(const uint4*)(p + 16);
      }
#pragma unroll
      for (int ct = 0; ct < CPW2; ++ct) {
        const int col = (wave * CPW2 + ct) * 16 + laneLo;
        const unsigned short* p = W2t + (size_t)col * HID + k0 + laneHi * 8;
        BF16Frag bfr;
        bfr.q[0] = *(const uint4*)(p);
        bfr.q[1] = *(const uint4*)(p + 16);
        __builtin_prefetch(p + 32, 0, 1);
#pragma unroll
        for (int rt = 0; rt < RT; ++rt)
          acc[rt][ct] = __builtin_amdgcn_wmma_f32_16x16x32_bf16(
              false, afr[rt].v, false, bfr.v, (short)0, acc[rt][ct], false, false);
      }
    }
#pragma unroll
    for (int ct = 0; ct < CPW2; ++ct) {
      const int col  = (wave * CPW2 + ct) * 16 + laneLo;
      const float bs = B2[col];
#pragma unroll
      for (int rt = 0; rt < RT; ++rt)
#pragma unroll
        for (int r = 0; r < 8; ++r) {
          const long long grow = rowbase + rt * 16 + r + laneHi * 8;
          const float o = acc[rt][ct][r] + bs;
          if (OUT_BF16)
            ((unsigned short*)outp)[grow * FOUT + col] = f2bf(o);
          else
            ((float*)outp)[grow * FOUT + col] = o;
        }
    }
  }
}

// ---------------------------------------------------------------------------
// Attention tail: per (n,a) block. q[256] f32, k/v[48][256] bf16.
//   logits -> mask bias -> softmax over 48 neighbors -> weighted sum of v.
//   out = a + attn_out.  (~0.15 GFLOP total: VALU is fine here.)
// ---------------------------------------------------------------------------
#define NEG_BIAS (-100000.0f)

__global__ __launch_bounds__(256) void attn_kernel(
    const float* __restrict__ qbuf, const unsigned short* __restrict__ kbuf,
    const unsigned short* __restrict__ vbuf, const int* __restrict__ mask,
    const float* __restrict__ a_in, float* __restrict__ out) {
  __shared__ float qs[256];
  __shared__ float attw[48 * 8];

  const int ra  = blockIdx.x;  // flattened (n, a), 0..3071
  const int tid = threadIdx.x;

  qs[tid] = qbuf[(size_t)ra * 256 + tid];
  __syncthreads();

  // logits: 48 neighbors x 8 heads, head dim 32
  for (int p = tid; p < 48 * 8; p += 256) {
    const int bi = p >> 3, h = p & 7;
    const unsigned short* kr = kbuf + ((size_t)ra * 48 + bi) * 256 + h * 32;
    float s = 0.f;
#pragma unroll
    for (int d = 0; d < 32; ++d) s += qs[h * 32 + d] * bf2f(kr[d]);
    if (mask[ra * 48 + bi] == 0) s += NEG_BIAS;
    attw[p] = s;
  }
  __syncthreads();

  // softmax over neighbors, one head per thread
  if (tid < 8) {
    const int h = tid;
    float m = -1e30f;
    for (int b = 0; b < 48; ++b) m = fmaxf(m, attw[b * 8 + h]);
    float s = 0.f;
    for (int b = 0; b < 48; ++b) {
      float e = __expf(attw[b * 8 + h] - m);
      attw[b * 8 + h] = e;
      s += e;
    }
    const float inv = 1.f / s;
    for (int b = 0; b < 48; ++b) attw[b * 8 + h] *= inv;
  }
  __syncthreads();

  // weighted sum of v + residual
  const int h = tid >> 5;
  const unsigned short* vr = vbuf + (size_t)ra * 48 * 256 + tid;
  float s = 0.f;
#pragma unroll 4
  for (int b = 0; b < 48; ++b) s += attw[b * 8 + h] * bf2f(vr[(size_t)b * 256]);
  out[(size_t)ra * 256 + tid] = a_in[(size_t)ra * 256 + tid] + s;
}

// ---------------------------------------------------------------------------
// Host launcher
// ---------------------------------------------------------------------------
extern "C" void kernel_launch(void* const* d_in, const int* in_sizes, int n_in,
                              void* d_out, int out_size, void* d_ws, size_t ws_size,
                              hipStream_t stream) {
  (void)in_sizes; (void)n_in; (void)out_size; (void)ws_size;

  const float* a_in = (const float*)d_in[0];   // [32,96,256]
  const float* b_in = (const float*)d_in[1];   // [32,96,48,256]
  const int*   mask = (const int*)d_in[2];     // [32,96,48]
  const float* rbf  = (const float*)d_in[3];   // [32,96,48,256]
  const float* q_w1 = (const float*)d_in[4];
  const float* q_b1 = (const float*)d_in[5];
  const float* q_w2 = (const float*)d_in[6];
  const float* q_b2 = (const float*)d_in[7];
  const float* k_w1 = (const float*)d_in[8];
  const float* k_b1 = (const float*)d_in[9];
  const float* k_w2 = (const float*)d_in[10];
  const float* k_b2 = (const float*)d_in[11];
  const float* v_w1 = (const float*)d_in[12];
  const float* v_b1 = (const float*)d_in[13];
  const float* v_w2 = (const float*)d_in[14];
  const float* v_b2 = (const float*)d_in[15];

  const long long ROWS_QA = 32LL * 96;            // 3072
  const long long ROWS_KV = 32LL * 96 * 48;       // 147456

  // workspace carve-up
  char*  ws  = (char*)d_ws;
  size_t off = 0;
  auto carve = [&](size_t bytes) {
    void* p = ws + off;
    off += (bytes + 255) & ~(size_t)255;
    return p;
  };
  unsigned short* Wq1 = (unsigned short*)carve((size_t)256 * 512 * 2);
  unsigned short* Wq2 = (unsigned short*)carve((size_t)512 * 256 * 2);
  unsigned short* Wk1 = (unsigned short*)carve((size_t)256 * 512 * 2);
  unsigned short* Wk2 = (unsigned short*)carve((size_t)512 * 256 * 2);
  unsigned short* Wv1 = (unsigned short*)carve((size_t)512 * 512 * 2);
  unsigned short* Wv2 = (unsigned short*)carve((size_t)512 * 256 * 2);
  float*          qbuf = (float*)carve((size_t)ROWS_QA * 256 * 4);
  unsigned short* kbuf = (unsigned short*)carve((size_t)ROWS_KV * 256 * 2);
  unsigned short* vbuf = (unsigned short*)carve((size_t)ROWS_KV * 256 * 2);

  // weight convert + transpose (tiny)
  wconv_kernel<<<(256 * 512 + 255) / 256, 256, 0, stream>>>(q_w1, Wq1, 256, 512);
  wconv_kernel<<<(512 * 256 + 255) / 256, 256, 0, stream>>>(q_w2, Wq2, 512, 256);
  wconv_kernel<<<(256 * 512 + 255) / 256, 256, 0, stream>>>(k_w1, Wk1, 256, 512);
  wconv_kernel<<<(512 * 256 + 255) / 256, 256, 0, stream>>>(k_w2, Wk2, 512, 256);
  wconv_kernel<<<(512 * 512 + 255) / 256, 256, 0, stream>>>(v_w1, Wv1, 512, 512);
  wconv_kernel<<<(512 * 256 + 255) / 256, 256, 0, stream>>>(v_w2, Wv2, 512, 256);

  const size_t lds_qk = (size_t)(64 * 256 + 64 * 512) * 2;  //  96 KB
  const size_t lds_v  = (size_t)(64 * 512 + 64 * 512) * 2;  // 128 KB (WGP has 320 KB)

  (void)hipFuncSetAttribute((const void*)mlp_kernel<256, 512, 256, false>,
                            hipFuncAttributeMaxDynamicSharedMemorySize, (int)lds_qk);
  (void)hipFuncSetAttribute((const void*)mlp_kernel<256, 512, 256, true>,
                            hipFuncAttributeMaxDynamicSharedMemorySize, (int)lds_qk);
  (void)hipFuncSetAttribute((const void*)mlp_kernel<512, 512, 256, true>,
                            hipFuncAttributeMaxDynamicSharedMemorySize, (int)lds_v);

  // q-MLP (f32 out), k-MLP (bf16 out), v-MLP on concat(b, rbf) (bf16 out)
  mlp_kernel<256, 512, 256, false><<<(int)(ROWS_QA / 64), 256, lds_qk, stream>>>(
      a_in, nullptr, 256, Wq1, q_b1, Wq2, q_b2, qbuf);
  mlp_kernel<256, 512, 256, true><<<(int)(ROWS_KV / 64), 256, lds_qk, stream>>>(
      b_in, nullptr, 256, Wk1, k_b1, Wk2, k_b2, kbuf);
  mlp_kernel<512, 512, 256, true><<<(int)(ROWS_KV / 64), 256, lds_v, stream>>>(
      b_in, rbf, 256, Wv1, v_b1, Wv2, v_b2, vbuf);

  // attention tail + residual
  attn_kernel<<<(int)ROWS_QA, 256, 0, stream>>>(qbuf, kbuf, vbuf, mask, a_in,
                                                (float*)d_out);
}
